// LGHIFusion_9517647528368
// MI455X (gfx1250) — matrix operand order repaired and checked by reading, not compiled, not run on hardware
//
#include <hip/hip_runtime.h>
#include <hip/hip_bf16.h>

// ---------------------------------------------------------------------------
// CDNA5 (gfx1250) fused attention block:
//   Q = low @ Wq^T + bq ; K = low @ Wk^T + bk ; V = high @ Wv^T + bv
//   ctx = softmax(Q K^T / sqrt(Dh)) V        (flash-attention, on-chip)
//   out = low + sigmoid(gamma) * (ctx @ Wo^T + bo)
// All GEMMs via v_wmma_f32_16x16x32_f16 (wave32), f32 accumulation.
// Tiles are laid out so every fragment load is a contiguous ds_load_b128 pair
// (ISA 7.12.2); global->LDS staging is batched for memory-level parallelism.
// GEMMs use 64-row x 128-col workgroup tiles: 8 WMMA per wave per k-step with
// 4x B-fragment reuse.
// ---------------------------------------------------------------------------

typedef __attribute__((ext_vector_type(4)))  _Float16 v4h;
typedef __attribute__((ext_vector_type(8)))  _Float16 v8h;
typedef __attribute__((ext_vector_type(16))) _Float16 v16h;
typedef __attribute__((ext_vector_type(8)))  float    v8f;

#define D_MODEL 1024
#define N_HEADS 16
#define D_HEAD  64
#define SEQ     2048
#define SEQ_SH  11          // log2(SEQ)

__device__ __forceinline__ v8f wmma_f16(v16h a, v16h b, v8f c) {
    // 8 args: (neg_a, A, neg_b, B, c_mod, C, reuse_a, reuse_b)
    return __builtin_amdgcn_wmma_f32_16x16x32_f16(
        false, a, false, b, (short)0, c, false, false);
}

__device__ __forceinline__ v16h cat16(v8h lo, v8h hi) {
    return __builtin_shufflevector(lo, hi, 0, 1, 2, 3, 4, 5, 6, 7,
                                           8, 9, 10, 11, 12, 13, 14, 15);
}

__device__ __forceinline__ v4h cvt4(float4 f) {
    v4h h;
    h[0] = (_Float16)f.x; h[1] = (_Float16)f.y;
    h[2] = (_Float16)f.z; h[3] = (_Float16)f.w;
    return h;
}

// A fragment (16x32 f16, MxK) from row-major row pointer (ld = 64 halves).
// Lane: M = lane&15, half = lane>>4.  Elements 0..7 -> k = kbase+half*8..+7,
// elements 8..15 -> k = kbase+16+half*8..+7: two contiguous b128 loads.
__device__ __forceinline__ v16h load_a_frag(const _Float16* row, int kbase,
                                            int half) {
    const v8h lo = *(const v8h*)(row + kbase + half * 8);
    const v8h hi = *(const v8h*)(row + kbase + 16 + half * 8);
    return cat16(lo, hi);
}

// B fragment (32x16 f16, KxN) from N-major row pointer (row = output column,
// ld = 64 halves).  Elements 0..7 -> k = kbase+half*8..+7, elements 8..15 ->
// k = kbase+16+half*8..+7: two contiguous b128 loads.
__device__ __forceinline__ v16h load_b_frag(const _Float16* nrow, int kbase,
                                            int half) {
    const v8h lo = *(const v8h*)(nrow + kbase + half * 8);
    const v8h hi = *(const v8h*)(nrow + kbase + 16 + half * 8);
    return cat16(lo, hi);
}

// ---------------------------------------------------------------------------
// Kernel 1: QKV projection.  grid = (B*S/64, D/128, 3), block = 256 (8 waves).
// Workgroup tile: 64 rows x 128 cols.  Wave w owns columns col0+16w and all
// four 16-row sub-tiles (B fragments reused 4x).  K-step = 64.
// Output layout: (B, H, S, Dh) f16.
// ---------------------------------------------------------------------------
__global__ void __launch_bounds__(256)
qkv_proj_kernel(const float* __restrict__ low, const float* __restrict__ high,
                const float* __restrict__ WQ, const float* __restrict__ bQ,
                const float* __restrict__ WK, const float* __restrict__ bK,
                const float* __restrict__ WV, const float* __restrict__ bV,
                _Float16* __restrict__ qw, _Float16* __restrict__ kw,
                _Float16* __restrict__ vw) {
    const float* X; const float* W; const float* bias; _Float16* O;
    switch (blockIdx.z) {
        case 0:  X = low;  W = WQ; bias = bQ; O = qw; break;
        case 1:  X = low;  W = WK; bias = bK; O = kw; break;
        default: X = high; W = WV; bias = bV; O = vw; break;
    }
    __shared__ _Float16 lds_a[64 * 64];     // x tile [row][k]
    __shared__ _Float16 lds_b[128 * 64];    // W tile [m_local][k]  (N-major)

    const int row0 = blockIdx.x * 64;
    const int col0 = blockIdx.y * 128;
    const int tid  = threadIdx.x;
    const int wave = tid >> 5;
    const int lane = tid & 31;
    const int m    = lane & 15;
    const int half = (lane >> 4) & 1;

    v8f acc[4];
    const v8f vzero = {};
#pragma unroll
    for (int s = 0; s < 4; ++s) acc[s] = vzero;

    for (int k0 = 0; k0 < D_MODEL; k0 += 64) {
        __syncthreads();
        // ---- batched global loads first (MLP), then convert + LDS store ----
        float4 abuf[4];   // A: 64 rows x 16 float4 = 1024 -> 4/thread
        float4 wbuf[8];   // B: 128 rows x 16 float4 = 2048 -> 8/thread
#pragma unroll
        for (int j = 0; j < 4; ++j) {
            const int i = tid + j * 256;
            const int rr = i >> 4, q = i & 15;
            abuf[j] = *(const float4*)&X[(size_t)(row0 + rr) * D_MODEL + k0 + 4 * q];
        }
#pragma unroll
        for (int j = 0; j < 8; ++j) {
            const int i = tid + j * 256;
            const int mm = i >> 4, q = i & 15;
            wbuf[j] = *(const float4*)&W[(size_t)(col0 + mm) * D_MODEL + k0 + 4 * q];
        }
#pragma unroll
        for (int j = 0; j < 4; ++j) {
            const int i = tid + j * 256;
            const int rr = i >> 4, q = i & 15;
            *(v4h*)&lds_a[rr * 64 + 4 * q] = cvt4(abuf[j]);
        }
#pragma unroll
        for (int j = 0; j < 8; ++j) {
            const int i = tid + j * 256;
            const int mm = i >> 4, q = i & 15;
            *(v4h*)&lds_b[mm * 64 + 4 * q] = cvt4(wbuf[j]);
        }
        if (k0 + 64 < D_MODEL)   // overlap next W tile fetch with this MMA
            __builtin_prefetch(
                &W[(size_t)(col0 + (tid >> 1)) * D_MODEL + k0 + 64], 0, 1);
        __syncthreads();

        const _Float16* brow = lds_b + (wave * 16 + m) * 64;
        const v16h b0 = load_b_frag(brow, 0,  half);
        const v16h b1 = load_b_frag(brow, 32, half);
#pragma unroll
        for (int s = 0; s < 4; ++s) {
            const _Float16* arow = lds_a + (s * 16 + m) * 64;
            acc[s] = wmma_f16(load_a_frag(arow, 0,  half), b0, acc[s]);
            acc[s] = wmma_f16(load_a_frag(arow, 32, half), b1, acc[s]);
        }
    }

    // C layout: col = col0 + 16*wave + (lane&15), row = sub*16 + r + 8*half
    const int f = col0 + wave * 16 + m;
    const int h = f >> 6, d = f & 63;
    const float bf = bias[f];
#pragma unroll
    for (int s = 0; s < 4; ++s)
#pragma unroll
        for (int r = 0; r < 8; ++r) {
            const unsigned tok = (unsigned)(row0 + s * 16 + r + half * 8);
            const unsigned bb  = tok >> SEQ_SH;
            const unsigned ss  = tok & (SEQ - 1);
            O[(((size_t)bb * N_HEADS + h) * SEQ + ss) * D_HEAD + d] =
                (_Float16)(acc[s][r] + bf);
        }
}

// ---------------------------------------------------------------------------
// Kernel 2: flash attention.  grid = (S/128, H, B), block = 256 (8 waves).
// Wave: 16 query rows; loop over 64-key blocks (16 WMMA each), online softmax.
// ---------------------------------------------------------------------------
__global__ void __launch_bounds__(256)
flash_attn_kernel(const _Float16* __restrict__ Qw,
                  const _Float16* __restrict__ Kw,
                  const _Float16* __restrict__ Vw,
                  _Float16* __restrict__ ctx) {
    const int bh = blockIdx.z * N_HEADS + blockIdx.y;
    const _Float16* Qp = Qw + (size_t)bh * SEQ * D_HEAD;
    const _Float16* Kp = Kw + (size_t)bh * SEQ * D_HEAD;
    const _Float16* Vp = Vw + (size_t)bh * SEQ * D_HEAD;

    const int tid  = threadIdx.x;
    const int wave = tid >> 5;
    const int lane = tid & 31;
    const int m    = lane & 15;
    const int half = (lane >> 4) & 1;
    const int q0   = blockIdx.x * 128 + wave * 16;

    __shared__ _Float16 lds_k[64 * 64];     // [key][d]  (N-major for scores)
    __shared__ _Float16 lds_vT[64 * 64];    // [d][key]  (N-major for P@V)
    __shared__ _Float16 lds_p[8][16 * 64];  // per-wave P tile [q][key]

    // Q fragments held in registers for the whole pass (4 x global b128/lane)
    const _Float16* qrow = Qp + (size_t)(q0 + m) * D_HEAD;
    const v16h qa0 = load_a_frag(qrow, 0,  half);
    const v16h qa1 = load_a_frag(qrow, 32, half);

    float rm[8], rl[8];
    v8f accd[4];
    const v8f vzero = {};
#pragma unroll
    for (int r = 0; r < 8; ++r) { rm[r] = -1e30f; rl[r] = 0.0f; }
#pragma unroll
    for (int t = 0; t < 4; ++t) accd[t] = vzero;

    const float scale = 0.125f;  // 1/sqrt(Dh)

    for (int kb = 0; kb < SEQ; kb += 64) {
        __syncthreads();
        // ---- batched loads (MLP), then LDS stores ----
        v8h kbuf[2], vbuf[2];
#pragma unroll
        for (int j = 0; j < 2; ++j) {    // 512 chunks of 8 halves, 2/thread
            const int i   = tid + j * 256;
            const int key = i >> 3, q = i & 7;
            kbuf[j] = *(const v8h*)&Kp[(size_t)(kb + key) * D_HEAD + 8 * q];
            vbuf[j] = *(const v8h*)&Vp[(size_t)(kb + key) * D_HEAD + 8 * q];
        }
#pragma unroll
        for (int j = 0; j < 2; ++j) {
            const int i   = tid + j * 256;
            const int key = i >> 3, q = i & 7;
            *(v8h*)&lds_k[key * 64 + 8 * q] = kbuf[j];
#pragma unroll
            for (int e = 0; e < 8; ++e)   // transpose V into [d][key]
                lds_vT[(8 * q + e) * 64 + key] = vbuf[j][e];
        }
        if (kb + 64 < SEQ) {
            __builtin_prefetch(&Kp[(size_t)(kb + 64 + (tid >> 2)) * D_HEAD], 0, 1);
            __builtin_prefetch(&Vp[(size_t)(kb + 64 + (tid >> 2)) * D_HEAD], 0, 1);
        }
        __syncthreads();

        // scores: Q(16x64) @ K^T(64x64) -> four 16x16 tiles, K split 2x32
        v8f sc[4];
#pragma unroll
        for (int nt = 0; nt < 4; ++nt) {
            const _Float16* krow = lds_k + (nt * 16 + m) * 64;
            v8f c = vzero;
            c = wmma_f16(qa0, load_b_frag(krow, 0,  half), c);
            c = wmma_f16(qa1, load_b_frag(krow, 32, half), c);
            sc[nt] = c;
        }

        // online softmax; row r + 8*half lives across 16 lanes of one half
        float alpha[8];
#pragma unroll
        for (int r = 0; r < 8; ++r) {
            const float v0 = sc[0][r] * scale;
            const float v1 = sc[1][r] * scale;
            const float v2 = sc[2][r] * scale;
            const float v3 = sc[3][r] * scale;
            float mx = fmaxf(fmaxf(v0, v1), fmaxf(v2, v3));
#pragma unroll
            for (int off = 8; off >= 1; off >>= 1)
                mx = fmaxf(mx, __shfl_xor(mx, off, 32));
            const float mnew = fmaxf(rm[r], mx);
            const float a  = __expf(rm[r] - mnew);
            const float p0 = __expf(v0 - mnew);
            const float p1 = __expf(v1 - mnew);
            const float p2 = __expf(v2 - mnew);
            const float p3 = __expf(v3 - mnew);
            float ps = (p0 + p1) + (p2 + p3);
#pragma unroll
            for (int off = 8; off >= 1; off >>= 1)
                ps += __shfl_xor(ps, off, 32);
            rl[r] = rl[r] * a + ps;
            rm[r] = mnew;
            alpha[r] = a;
            _Float16* prow = &lds_p[wave][(r + half * 8) * 64];
            prow[m]      = (_Float16)p0;
            prow[16 + m] = (_Float16)p1;
            prow[32 + m] = (_Float16)p2;
            prow[48 + m] = (_Float16)p3;
        }

#pragma unroll
        for (int t = 0; t < 4; ++t)
#pragma unroll
            for (int r = 0; r < 8; ++r) accd[t][r] *= alpha[r];

        // ctx += P(16x64) @ V(64x64): four 16x16 N-tiles, K split 2x32
        const _Float16* parow = &lds_p[wave][m * 64];
        const v16h pa0 = load_a_frag(parow, 0,  half);
        const v16h pa1 = load_a_frag(parow, 32, half);
#pragma unroll
        for (int t = 0; t < 4; ++t) {
            const _Float16* vrow = lds_vT + (t * 16 + m) * 64;
            accd[t] = wmma_f16(pa0, load_b_frag(vrow, 0,  half), accd[t]);
            accd[t] = wmma_f16(pa1, load_b_frag(vrow, 32, half), accd[t]);
        }
    }

    // finalize: divide by row sum, write ctx in (B,S,D) f16
    const int bz = blockIdx.z, hh = blockIdx.y;
    float inv_l[8];
#pragma unroll
    for (int r = 0; r < 8; ++r) inv_l[r] = 1.0f / rl[r];
#pragma unroll
    for (int t = 0; t < 4; ++t)
#pragma unroll
        for (int r = 0; r < 8; ++r) {
            const int s = q0 + r + half * 8;
            ctx[((size_t)bz * SEQ + s) * D_MODEL + hh * D_HEAD + t * 16 + m] =
                (_Float16)(accd[t][r] * inv_l[r]);
        }
}

// ---------------------------------------------------------------------------
// Kernel 3: output projection + gated residual.
// grid = (B*S/64, D/128), block = 256.  out = low + sigmoid(gamma)*(ctx Wo^T + bo)
// ---------------------------------------------------------------------------
__global__ void __launch_bounds__(256)
out_proj_kernel(const _Float16* __restrict__ ctx,
                const float* __restrict__ WO, const float* __restrict__ bO,
                const float* __restrict__ low, const float* __restrict__ gamma,
                float* __restrict__ out) {
    __shared__ _Float16 lds_a[64 * 64];
    __shared__ _Float16 lds_b[128 * 64];

    const int row0 = blockIdx.x * 64;
    const int col0 = blockIdx.y * 128;
    const int tid  = threadIdx.x;
    const int wave = tid >> 5;
    const int lane = tid & 31;
    const int m    = lane & 15;
    const int half = (lane >> 4) & 1;

    v8f acc[4];
    const v8f vzero = {};
#pragma unroll
    for (int s = 0; s < 4; ++s) acc[s] = vzero;

    for (int k0 = 0; k0 < D_MODEL; k0 += 64) {
        __syncthreads();
        // ---- batched loads (MLP), then LDS stores ----
        v8h   abuf[2];    // A (f16): 64 rows x 8 chunks = 512 -> 2/thread
        float4 wbuf[8];   // B: 128 rows x 16 float4 = 2048 -> 8/thread
#pragma unroll
        for (int j = 0; j < 2; ++j) {
            const int i = tid + j * 256;
            const int rr = i >> 3, q = i & 7;
            abuf[j] = *(const v8h*)&ctx[(size_t)(row0 + rr) * D_MODEL + k0 + 8 * q];
        }
#pragma unroll
        for (int j = 0; j < 8; ++j) {
            const int i = tid + j * 256;
            const int mm = i >> 4, q = i & 15;
            wbuf[j] = *(const float4*)&WO[(size_t)(col0 + mm) * D_MODEL + k0 + 4 * q];
        }
#pragma unroll
        for (int j = 0; j < 2; ++j) {
            const int i = tid + j * 256;
            const int rr = i >> 3, q = i & 7;
            *(v8h*)&lds_a[rr * 64 + 8 * q] = abuf[j];
        }
#pragma unroll
        for (int j = 0; j < 8; ++j) {
            const int i = tid + j * 256;
            const int mm = i >> 4, q = i & 15;
            *(v4h*)&lds_b[mm * 64 + 4 * q] = cvt4(wbuf[j]);
        }
        if (k0 + 64 < D_MODEL)
            __builtin_prefetch(
                &WO[(size_t)(col0 + (tid >> 1)) * D_MODEL + k0 + 64], 0, 1);
        __syncthreads();

        const _Float16* brow = lds_b + (wave * 16 + m) * 64;
        const v16h b0 = load_b_frag(brow, 0,  half);
        const v16h b1 = load_b_frag(brow, 32, half);
#pragma unroll
        for (int s = 0; s < 4; ++s) {
            const _Float16* arow = lds_a + (s * 16 + m) * 64;
            acc[s] = wmma_f16(load_a_frag(arow, 0,  half), b0, acc[s]);
            acc[s] = wmma_f16(load_a_frag(arow, 32, half), b1, acc[s]);
        }
    }

    const float beta = 1.0f / (1.0f + __expf(-gamma[0]));
    const int f = col0 + wave * 16 + m;
    const float bf = bO[f];
#pragma unroll
    for (int s = 0; s < 4; ++s)
#pragma unroll
        for (int r = 0; r < 8; ++r) {
            const unsigned tok = (unsigned)(row0 + s * 16 + r + half * 8);
            const size_t idx = (size_t)tok * D_MODEL + f;
            out[idx] = low[idx] + beta * (acc[s][r] + bf);
        }
}

// ---------------------------------------------------------------------------
extern "C" void kernel_launch(void* const* d_in, const int* in_sizes, int n_in,
                              void* d_out, int out_size, void* d_ws,
                              size_t ws_size, hipStream_t stream) {
    const float* low   = (const float*)d_in[0];
    const float* high  = (const float*)d_in[1];
    const float* WQ    = (const float*)d_in[2];
    const float* bQ    = (const float*)d_in[3];
    const float* WK    = (const float*)d_in[4];
    const float* bK    = (const float*)d_in[5];
    const float* WV    = (const float*)d_in[6];
    const float* bV    = (const float*)d_in[7];
    const float* WO    = (const float*)d_in[8];
    const float* bO    = (const float*)d_in[9];
    const float* gamma = (const float*)d_in[10];
    float* out = (float*)d_out;

    const int total  = in_sizes[0];              // B*S*D
    const int B      = total / (SEQ * D_MODEL);  // = 2
    const int n_rows = B * SEQ;                  // = 4096

    const size_t per = (size_t)n_rows * D_MODEL; // f16 elements per tensor
    _Float16* qw = (_Float16*)d_ws;
    _Float16* kw = qw + per;
    _Float16* vw = kw + per;
    _Float16* cw = vw + per;                     // 4 * 8 MB = 32 MB of ws

    dim3 g1(n_rows / 64, D_MODEL / 128, 3);
    qkv_proj_kernel<<<g1, 256, 0, stream>>>(low, high, WQ, bQ, WK, bK, WV, bV,
                                            qw, kw, vw);

    dim3 g2(SEQ / 128, N_HEADS, B);
    flash_attn_kernel<<<g2, 256, 0, stream>>>(qw, kw, vw, cw);

    dim3 g3(n_rows / 64, D_MODEL / 128, 1);
    out_proj_kernel<<<g3, 256, 0, stream>>>(cw, WO, bO, low, gamma, out);
}
